// AIHWKITSNN_12910671692096
// MI455X (gfx1250) — matrix operand order repaired and checked by reading, not compiled, not run on hardware
//
#include <hip/hip_runtime.h>
#include <hip/hip_bf16.h>
#include <stdint.h>

// ---------------- CDNA5 WMMA vector types ----------------
typedef __attribute__((ext_vector_type(16))) __bf16 v16bf;
typedef __attribute__((ext_vector_type(8)))  float  v8f;

#define T_STEPS 25
#define BATCH   4096
#define NIN     784
#define HID     500
#define NOUT    10
#define KPAD    800               // NIN padded to multiple of 32 (WMMA K)
#define KC      (KPAD/32)         // 25 K-chunks
#define NPAD    512               // HID padded to multiple of 16
#define NT      (NPAD/16)         // 32 N-tiles
#define ROWS    16                // batch rows per block (one M tile)
#define THREADS 256               // 8 wave32 waves
#define BETA    0.9f

// ---------------- JAX threefry2x32-20, key = jax.random.key(42) = [0,42] ----
__device__ __forceinline__ uint32_t rotl32(uint32_t v, int d) {
    return (v << d) | (v >> (32 - d));
}

__device__ __forceinline__ uint2 threefry2x32_42(uint32_t x0, uint32_t x1) {
    const uint32_t k0 = 0u, k1 = 42u;
    const uint32_t k2 = 0x1BD11BDAu ^ k0 ^ k1;
    x0 += k0; x1 += k1;
#define TF_R4(a,b,c,d)                                        \
    x0 += x1; x1 = rotl32(x1,a); x1 ^= x0;                    \
    x0 += x1; x1 = rotl32(x1,b); x1 ^= x0;                    \
    x0 += x1; x1 = rotl32(x1,c); x1 ^= x0;                    \
    x0 += x1; x1 = rotl32(x1,d); x1 ^= x0;
    TF_R4(13,15,26,6);  x0 += k1; x1 += k2 + 1u;
    TF_R4(17,29,16,24); x0 += k2; x1 += k0 + 2u;
    TF_R4(13,15,26,6);  x0 += k0; x1 += k1 + 3u;
    TF_R4(17,29,16,24); x0 += k1; x1 += k2 + 4u;
    TF_R4(13,15,26,6);  x0 += k2; x1 += k0 + 5u;
#undef TF_R4
    return make_uint2(x0, x1);
}

// ---------------- pack W1 (f32, 784x500) -> bf16 B-fragment layout ----------
// Fragment (nt, kc): 32 lanes x 16 bf16 (1 KB), lane = n%16 + 16*(kk>=16),
// pos = kk%16  (contiguous k order, same as the A packing of spikes).
__global__ __launch_bounds__(256)
void pack_w1_kernel(const float* __restrict__ W1, unsigned short* __restrict__ W1p)
{
    int idx = blockIdx.x * 256 + threadIdx.x;
    if (idx >= KPAD * NPAD) return;
    int k = idx / NPAD, n = idx - k * NPAD;
    float v = (k < NIN && n < HID) ? W1[k * HID + n] : 0.0f;
    uint32_t u = __float_as_uint(v);
    u += 0x7FFFu + ((u >> 16) & 1u);               // round-to-nearest-even
    unsigned short h = (unsigned short)(u >> 16);
    int kc = k >> 5, kk = k & 31, nt = n >> 4, nn = n & 15;
    int ln  = nn | ((kk >> 4) << 4);
    int pos = kk & 15;
    W1p[(nt * KC + kc) * 512 + ln * 16 + pos] = h;
}

// ---------------- fused persistent SNN kernel ------------------------------
__global__ __launch_bounds__(THREADS)
void snn_fused_kernel(const float* __restrict__ x,
                      const unsigned short* __restrict__ W1p,
                      const float* __restrict__ b1,
                      const float* __restrict__ W2,
                      const float* __restrict__ b2,
                      float* __restrict__ out)
{
    // spikes in A-fragment layout: KC frags x 32 lanes x 16 bf16 = 25,600 B
    __shared__ __align__(64) unsigned short sSpk[KC * 512];
    // layer-1 spikes, row-major bf16 [16][512] = 16,384 B
    __shared__ __align__(64) unsigned short sSpk1[ROWS * NPAD];
    __shared__ float sW2[HID * NOUT];                    // 20,000 B
    __shared__ float sB2[NOUT];

    const int tid  = threadIdx.x;
    const int lane = tid & 31;
    const int wv   = tid >> 5;
    const int r0   = blockIdx.x * ROWS;

    for (int i = tid; i < HID * NOUT; i += THREADS) sW2[i] = W2[i];
    if (tid < NOUT) sB2[tid] = b2[tid];

    // per-wave bias / column bookkeeping for 4 N-tiles
    float bias[4];
    int   cols[4];
#pragma unroll
    for (int q = 0; q < 4; ++q) {
        int col = (wv * 4 + q) * 16 + (lane & 15);
        cols[q] = col;
        bias[q] = (col < HID) ? b1[col] : 0.0f;
    }

    // persistent membrane state: mem1 in WMMA C-layout registers, mem2 per thread
    float m1v[4][8];
#pragma unroll
    for (int q = 0; q < 4; ++q)
#pragma unroll
        for (int j = 0; j < 8; ++j) m1v[q][j] = 0.0f;
    float m2r = 0.0f;
    const int rr = tid / NOUT, oo = tid - rr * NOUT;     // layer-2 (row,out)

    __syncthreads();

    for (int t = 0; t < T_STEPS; ++t) {
        // ---- spike generation (exact JAX threefry split-counter scheme) ----
        for (int e = tid; e < ROWS * KPAD; e += THREADS) {
            int rl = e / KPAD, c = e - rl * KPAD;
            unsigned short s = 0;
            if (c < NIN) {
                uint32_t gidx  = (uint32_t)(t * (BATCH * NIN)) +
                                 (uint32_t)((r0 + rl) * NIN + c);
                const uint32_t HALFN = (uint32_t)(T_STEPS) * BATCH * NIN / 2u;
                uint32_t bits;
                if (gidx < HALFN) { uint2 o = threefry2x32_42(gidx, gidx + HALFN); bits = o.x; }
                else              { uint2 o = threefry2x32_42(gidx - HALFN, gidx); bits = o.y; }
                float u = __uint_as_float((bits >> 9) | 0x3F800000u) - 1.0f;
                float rate = x[(r0 + rl) * NIN + c];
                rate = fminf(fmaxf(rate, 0.0f), 1.0f);
                if (u < rate) s = 0x3F80;                // bf16 1.0
            }
            // A fragment, contiguous k order: lane = m + 16*(kk>=16), pos = kk%16 (as W1p)
            int kc  = c >> 5, kk = c & 31;
            int ln  = (rl & 15) | ((kk >> 4) << 4);
            int pos = kk & 15;
            sSpk[kc * 512 + ln * 16 + pos] = s;
        }
        __syncthreads();

        // ---- layer 1: [16 x 800] x [800 x 512] via v_wmma_f32_16x16x32_bf16
        v8f acc0 = {0.f,0.f,0.f,0.f,0.f,0.f,0.f,0.f};
        v8f acc[4] = {acc0, acc0, acc0, acc0};
        for (int kc = 0; kc < KC; ++kc) {
            v16bf a = *reinterpret_cast<const v16bf*>(&sSpk[kc * 512 + lane * 16]);
#pragma unroll
            for (int q = 0; q < 4; ++q) {
                int nt = wv * 4 + q;
                v16bf bf = *reinterpret_cast<const v16bf*>(
                    &W1p[(nt * KC + kc) * 512 + lane * 16]);
                acc[q] = __builtin_amdgcn_wmma_f32_16x16x32_bf16(
                    false, a, false, bf, (short)0, acc[q], false, false);
            }
            if (kc + 1 < KC)
                __builtin_prefetch(&W1p[((wv * 4) * KC + kc + 1) * 512 + lane * 16], 0, 1);
        }

        // ---- layer-1 LIF (reset on pre-update mem), spk1 -> LDS ----
        int rbase = (lane >> 4) << 3;                    // C layout: M = j + 8*(lane>=16)
#pragma unroll
        for (int q = 0; q < 4; ++q) {
#pragma unroll
            for (int j = 0; j < 8; ++j) {
                float cur = acc[q][j] + bias[q];
                float m   = m1v[q][j];
                float rst = (m > 1.0f) ? 1.0f : 0.0f;
                m = BETA * m + cur - rst;
                m1v[q][j] = m;
                sSpk1[(rbase + j) * NPAD + cols[q]] =
                    (m > 1.0f) ? (unsigned short)0x3F80 : (unsigned short)0;
            }
        }
        __syncthreads();

        // ---- layer 2: 500 -> 10 dot products + LIF + output ----
        if (tid < ROWS * NOUT) {
            float a2 = sB2[oo];
            for (int h = 0; h < HID; ++h) {
                float s1 = __uint_as_float(((uint32_t)sSpk1[rr * NPAD + h]) << 16);
                a2 += s1 * sW2[h * NOUT + oo];
            }
            float m   = m2r;
            float rst = (m > 1.0f) ? 1.0f : 0.0f;
            m = BETA * m + a2 - rst;
            m2r = m;
            out[((size_t)t * BATCH + (r0 + rr)) * NOUT + oo] = (m > 1.0f) ? 1.0f : 0.0f;
        }
        __syncthreads();
    }
}

// ---------------- launch ----------------------------------------------------
extern "C" void kernel_launch(void* const* d_in, const int* in_sizes, int n_in,
                              void* d_out, int out_size, void* d_ws, size_t ws_size,
                              hipStream_t stream) {
    (void)in_sizes; (void)n_in; (void)out_size; (void)ws_size;
    const float* x  = (const float*)d_in[0];
    const float* W1 = (const float*)d_in[1];
    const float* b1 = (const float*)d_in[2];
    const float* W2 = (const float*)d_in[3];
    const float* b2 = (const float*)d_in[4];
    unsigned short* W1p = (unsigned short*)d_ws;   // 800 KB packed bf16 W1

    int total = KPAD * NPAD;
    pack_w1_kernel<<<(total + 255) / 256, 256, 0, stream>>>(W1, W1p);
    snn_fused_kernel<<<BATCH / ROWS, THREADS, 0, stream>>>(
        x, W1p, b1, W2, b2, (float*)d_out);
}
